// SimpleS4Wrapper_20993800142894
// MI455X (gfx1250) — matrix operand order repaired and checked by reading, not compiled, not run on hardware
//
#include <hip/hip_runtime.h>
#include <hip/hip_bf16.h>
#include <cstdint>
#include <cstddef>

// Problem constants (reference: B=8, H=768, L=2048, C=1, DHALF=32)
#define S4_B 8
#define S4_H 768
#define S4_L 2048
#define S4_D 32
#define S4_N (S4_B * S4_L)   // 16384 = GEMM N
#define GM S4_H              // 768
#define GN S4_N              // 16384
#define GK S4_H              // 768
#define BM 128
#define BN 128
#define BK 32
#define KT (GK / BK)         // 24

typedef __attribute__((ext_vector_type(16))) __bf16 v16bf;
typedef __attribute__((ext_vector_type(8)))  __bf16 v8bf;
typedef __attribute__((ext_vector_type(8)))  float  v8f;
typedef __attribute__((ext_vector_type(8)))  unsigned short v8us;
typedef int v4i_ __attribute__((vector_size(4 * sizeof(int))));

#define S4_AS1 __attribute__((address_space(1)))
#define S4_AS3 __attribute__((address_space(3)))

#if defined(__gfx1250__) && __has_builtin(__builtin_amdgcn_global_load_async_to_lds_b128) && __has_builtin(__builtin_amdgcn_s_wait_asynccnt)
#define S4_ASYNC 1
#else
#define S4_ASYNC 0
#endif

static __device__ __forceinline__ unsigned short f2bf(float x) {
  unsigned int u = __float_as_uint(x);
  u += 0x7FFFu + ((u >> 16) & 1u);   // round-to-nearest-even
  return (unsigned short)(u >> 16);
}

// ---------------------------------------------------------------------------
// Kernel 1: W_out f32 -> bf16
// ---------------------------------------------------------------------------
__global__ __launch_bounds__(256) void s4_cvt_w(const float* __restrict__ W,
                                                unsigned short* __restrict__ Wb,
                                                int n) {
  int i = blockIdx.x * 256 + threadIdx.x;
  if (i < n) Wb[i] = f2bf(W[i]);
}

// ---------------------------------------------------------------------------
// Kernel 2: S4 recurrence + D*u + x0 bias + erf-GELU -> bf16 activations
// One wave32 per (b,h); lane = complex state d (DHALF==32 == wave size).
// ---------------------------------------------------------------------------
__global__ __launch_bounds__(256) void s4_scan(const float* __restrict__ u,
                                               const float* __restrict__ theta,
                                               const float* __restrict__ a,
                                               const float* __restrict__ bq,
                                               const float* __restrict__ cq,
                                               const float* __restrict__ x0,
                                               const float* __restrict__ Dp,
                                               unsigned short* __restrict__ yg) {
  __shared__ float red[8][S4_D * 32];           // per-wave 32x32 transpose pad: 32KB
  const int tid  = threadIdx.x;
  const int w    = tid >> 5;
  const int lane = tid & 31;                    // == d
  const int seq  = blockIdx.x * 8 + w;          // 0 .. B*H-1
  const int b = seq / S4_H;
  const int h = seq % S4_H;
  const float T = 1.0f / (float)(S4_L - 1);

  const int pd = h * S4_D + lane;
  const float th  = theta[pd];
  const float aa  = fabsf(a[pd]);
  const float bqv = bq[pd];
  const float cqv = cq[pd];
  const float x0v = x0[pd];
  const float Dh  = Dp[h];

  // lambda = exp((-aa + i*th) * T)
  const float r      = expf(-aa * T);
  const float lam_re = r * cosf(th * T);
  const float lam_im = r * sinf(th * T);
  // term0 = (lambda - 1) / (-aa + i*th)
  const float den = aa * aa + th * th;
  const float xr = lam_re - 1.0f, xi = lam_im;
  const float t0_re = (xr * (-aa) + xi * th) / den;
  const float t0_im = (xi * (-aa) - xr * th) / den;
  // w = 2 * bq*cq * term0
  const float w_re = 2.0f * bqv * cqv * t0_re;
  const float w_im = 2.0f * bqv * cqv * t0_im;

  // bias geometric sequence: Re(4T*cq*x0 * lambda^l)
  float z_re = 4.0f * T * cqv * x0v;
  float z_im = 0.0f;
  float s_re = 0.0f, s_im = 0.0f;

  const float* ub = u + ((size_t)b * S4_H + h) * S4_L;
  unsigned short* yb = yg + (size_t)h * S4_N + (size_t)b * S4_L;
  float* myred = red[w];

  for (int l0 = 0; l0 < S4_L; l0 += 32) {
    const float uval = ub[l0 + lane];
    #pragma unroll
    for (int t = 0; t < 32; ++t) {
      const float us = __shfl(uval, t, 32);
      // s = lambda*s + u[l]
      const float nre = lam_re * s_re - lam_im * s_im + us;
      const float nim = lam_re * s_im + lam_im * s_re;
      s_re = nre; s_im = nim;
      // contribution of this state + bias sequence (before z advance)
      const float contrib = (w_re * s_re - w_im * s_im) + z_re;
      const float zr = lam_re * z_re - lam_im * z_im;
      const float zi = lam_re * z_im + lam_im * z_re;
      z_re = zr; z_im = zi;
      myred[t * 32 + lane] = contrib;
    }
    __syncthreads();
    // lane j reduces timestep l0+j over the 32 states
    float tot = 0.0f;
    #pragma unroll
    for (int dd = 0; dd < 32; ++dd) tot += myred[lane * 32 + dd];
    tot += Dh * uval;
    // exact erf-form GELU
    const float g = 0.5f * tot * (1.0f + erff(tot * 0.70710678118654752440f));
    yb[l0 + lane] = f2bf(g);
    __syncthreads();
  }
}

// ---------------------------------------------------------------------------
// Kernel 3: out[b,o,l] = sum_c W[o,c]*y[c,b*L+l] + b_out[o]
// bf16 WMMA GEMM, 128x128x32 tiles, 8 waves, 4x2 16x16 accumulators per wave,
// double-buffered LDS (B stored transposed [n][k] so fragments are b128 loads).
// A tile is staged with global_load_async_to_lds_b128 (ASYNCcnt-tracked).
// ---------------------------------------------------------------------------
__global__ __launch_bounds__(256) void s4_gemm(const unsigned short* __restrict__ Wb,
                                               const unsigned short* __restrict__ Yb,
                                               const float* __restrict__ bout,
                                               float* __restrict__ out) {
  __shared__ __align__(16) unsigned short sA[2][BM * BK];  // [m][k]   16KB
  __shared__ __align__(16) unsigned short sB[2][BN * BK];  // [n][k]^T 16KB

  const int tid  = threadIdx.x;
  const int lane = tid & 31;
  const int wv   = tid >> 5;
  const int wr   = wv >> 2;           // 0..1  -> 64-row slab
  const int wc   = wv & 3;            // 0..3  -> 32-col slab
  const int m_lane = lane & 15;
  const int gsel   = lane >> 4;
  const int m0 = blockIdx.y * BM;
  const int n0 = blockIdx.x * BN;

  v8f acc[4][2];
  #pragma unroll
  for (int mt = 0; mt < 4; ++mt)
    #pragma unroll
    for (int nt = 0; nt < 2; ++nt)
      #pragma unroll
      for (int i = 0; i < 8; ++i) acc[mt][nt][i] = 0.0f;

  // ---- prologue: stage k-tile 0 into buffer 0 (plain path) ----
  #pragma unroll
  for (int pass = 0; pass < 2; ++pass) {
    const int ch   = tid + pass * 256;
    const int arow = ch >> 2, acol = (ch & 3) * 8;
    const int krow = ch >> 4, nn = (ch & 15) * 8;
    v8us av = *(const v8us*)&Wb[(size_t)(m0 + arow) * GK + acol];
    *(v8us*)&sA[0][arow * BK + acol] = av;
    v8us bv = *(const v8us*)&Yb[(size_t)krow * GN + n0 + nn];
    #pragma unroll
    for (int i = 0; i < 8; ++i) sB[0][(nn + i) * BK + krow] = bv[i];
  }

  for (int kt = 0; kt < KT; ++kt) {
    const int cur = kt & 1;
    const int nxt = cur ^ 1;
#if S4_ASYNC
    __builtin_amdgcn_s_wait_asynccnt(0);
#endif
    __syncthreads();

    const bool has_next = (kt + 1) < KT;
    v8us bvr[2];
#if !S4_ASYNC
    v8us avr[2];
#endif
    if (has_next) {
      const int kb = (kt + 1) * BK;
      #pragma unroll
      for (int pass = 0; pass < 2; ++pass) {
        const int ch   = tid + pass * 256;
        const int arow = ch >> 2, acol = (ch & 3) * 8;
        const int krow = ch >> 4, nn = (ch & 15) * 8;
#if S4_ASYNC
        __builtin_amdgcn_global_load_async_to_lds_b128(
            (S4_AS1 v4i_*)&Wb[(size_t)(m0 + arow) * GK + kb + acol],
            (S4_AS3 v4i_*)&sA[nxt][arow * BK + acol],
            0, 0);
#else
        avr[pass] = *(const v8us*)&Wb[(size_t)(m0 + arow) * GK + kb + acol];
#endif
        bvr[pass] = *(const v8us*)&Yb[(size_t)(kb + krow) * GN + n0 + nn];
      }
      if (kt + 2 < KT) {
        const int kb2 = (kt + 2) * BK;
        __builtin_prefetch(&Wb[(size_t)(m0 + (tid >> 2)) * GK + kb2 + (tid & 3) * 8], 0, 1);
        __builtin_prefetch(&Yb[(size_t)(kb2 + (tid >> 4)) * GN + n0 + (tid & 15) * 8], 0, 1);
      }
    }

    // ---- compute on buffer `cur` ----
    // A fragment (16x32 bf16, M x K): lane m=lane%16, g=lane/16 holds
    //   K = g*8 + 0..7 (elems 0..7) and K = 16 + g*8 + 0..7 (elems 8..15)
    v16bf af[4];
    #pragma unroll
    for (int mt = 0; mt < 4; ++mt) {
      const int row = wr * 64 + mt * 16 + m_lane;
      v8bf lo = *(const v8bf*)&sA[cur][row * BK + gsel * 8];
      v8bf hi = *(const v8bf*)&sA[cur][row * BK + 16 + gsel * 8];
      af[mt] = __builtin_shufflevector(lo, hi, 0,1,2,3,4,5,6,7,8,9,10,11,12,13,14,15);
    }
    // B fragment (32x16 bf16, K x N): lane n=lane%16, g=lane/16 holds K = g*8 + 0..7 and K = 16 + g*8 + 0..7
    v16bf bfm[2];
    #pragma unroll
    for (int nt = 0; nt < 2; ++nt) {
      const int row = wc * 32 + nt * 16 + m_lane;   // transposed: row = n
      v8bf lo = *(const v8bf*)&sB[cur][row * BK + gsel * 8];
      v8bf hi = *(const v8bf*)&sB[cur][row * BK + 16 + gsel * 8];
      bfm[nt] = __builtin_shufflevector(lo, hi, 0,1,2,3,4,5,6,7,8,9,10,11,12,13,14,15);
    }
    #pragma unroll
    for (int mt = 0; mt < 4; ++mt)
      #pragma unroll
      for (int nt = 0; nt < 2; ++nt)
        acc[mt][nt] = __builtin_amdgcn_wmma_f32_16x16x32_bf16(
            false, af[mt], false, bfm[nt], (short)0, acc[mt][nt], false, false);

    // ---- commit staged tile into buffer `nxt` ----
    if (has_next) {
      #pragma unroll
      for (int pass = 0; pass < 2; ++pass) {
        const int ch = tid + pass * 256;
#if !S4_ASYNC
        const int arow = ch >> 2, acol = (ch & 3) * 8;
        *(v8us*)&sA[nxt][arow * BK + acol] = avr[pass];
#endif
        const int krow = ch >> 4, nn = (ch & 15) * 8;
        #pragma unroll
        for (int i = 0; i < 8; ++i) sB[nxt][(nn + i) * BK + krow] = bvr[pass][i];
      }
    }
  }

  // ---- epilogue: D layout -> element r of acc is (M = r + 8*gsel, N = lane%16)
  const int bI = n0 / S4_L;       // BN divides L, so each tile lies in one batch
  const int l0 = n0 % S4_L;
  #pragma unroll
  for (int mt = 0; mt < 4; ++mt) {
    #pragma unroll
    for (int nt = 0; nt < 2; ++nt) {
      const int ncol = l0 + wc * 32 + nt * 16 + m_lane;
      #pragma unroll
      for (int rr = 0; rr < 8; ++rr) {
        const int o = m0 + wr * 64 + mt * 16 + gsel * 8 + rr;
        out[((size_t)bI * GM + o) * S4_L + ncol] = acc[mt][nt][rr] + bout[o];
      }
    }
  }
}

// ---------------------------------------------------------------------------
extern "C" void kernel_launch(void* const* d_in, const int* in_sizes, int n_in,
                              void* d_out, int out_size, void* d_ws, size_t ws_size,
                              hipStream_t stream) {
  (void)in_sizes; (void)n_in; (void)out_size; (void)ws_size;
  const float* u     = (const float*)d_in[0];
  const float* theta = (const float*)d_in[1];
  const float* a     = (const float*)d_in[2];
  const float* bq    = (const float*)d_in[3];
  const float* cq    = (const float*)d_in[4];
  const float* x0    = (const float*)d_in[5];
  const float* Dp    = (const float*)d_in[6];
  const float* Wout  = (const float*)d_in[7];
  const float* bout  = (const float*)d_in[8];
  float* out = (float*)d_out;

  // Workspace: Wb (768*768 bf16 = 1.125 MB), then Yb (768*16384 bf16 = 24 MB)
  unsigned short* Wb = (unsigned short*)d_ws;
  unsigned short* Yb = (unsigned short*)((char*)d_ws + (size_t)GM * GK * 2);

  s4_cvt_w<<<(GM * GK + 255) / 256, 256, 0, stream>>>(Wout, Wb, GM * GK);
  s4_scan<<<(S4_B * S4_H) / 8, 256, 0, stream>>>(u, theta, a, bq, cq, x0, Dp, Yb);
  s4_gemm<<<dim3(GN / BN, GM / BM), 256, 0, stream>>>(Wb, Yb, bout, out);
}
